// FlattenedWindowMapping_214748365270
// MI455X (gfx1250) — matrix operand order repaired and checked by reading, not compiled, not run on hardware
//
#include <hip/hip_runtime.h>

typedef unsigned int u32;
typedef __attribute__((ext_vector_type(16))) _Float16 v16h;
typedef __attribute__((ext_vector_type(8)))  float    v8f;
typedef __attribute__((address_space(3))) u32 lds3_u32;

#define GSZ   128
#define NB    8
#define WX    12
#define WY    12
#define WZ    32
#define VOL   (WX*WY*WZ)
#define TILE  2048

// ws header layout (u32):
// [0..7]   counts per batch
// [8..16]  bsi   (exclusive prefix, 9 entries)
// [17..25] bsi_p (padded exclusive prefix, 9 entries)
// [26] my*mz  [27] mx*mz  [28] mz  [29] mper  [30] M_p (=bsi_p[8])

__global__ void k_setup(const int* __restrict__ sparse, u32* __restrict__ hdr) {
    if (threadIdx.x == 0 && blockIdx.x == 0) {
        int sz = sparse[0], sy = sparse[1], sx = sparse[2];
        u32 mx = (u32)((sx + WX - 1) / WX + 1);
        u32 my = (u32)((sy + WY - 1) / WY + 1);
        u32 mz = (u32)((sz + WZ - 1) / WZ + 1);
        hdr[26] = my * mz;
        hdr[27] = mx * mz;
        hdr[28] = mz;
        hdr[29] = mx * my * mz;
        for (int i = 0; i < NB; ++i) hdr[i] = 0;
    }
}

__global__ void k_count(const int* __restrict__ coords, int n, u32* __restrict__ hdr) {
    __shared__ u32 h[NB];
    if (threadIdx.x < NB) h[threadIdx.x] = 0;
    __syncthreads();
    for (int i = blockIdx.x * blockDim.x + threadIdx.x; i < n; i += gridDim.x * blockDim.x) {
        atomicAdd(&h[coords[4 * i] & (NB - 1)], 1u);
    }
    __syncthreads();
    if (threadIdx.x < NB) atomicAdd(&hdr[threadIdx.x], h[threadIdx.x]);
}

__global__ void k_offsets(u32* __restrict__ hdr) {
    if (threadIdx.x == 0 && blockIdx.x == 0) {
        u32 s = 0, sp = 0;
        hdr[8] = 0; hdr[17] = 0;
        for (int i = 0; i < NB; ++i) {
            u32 c  = hdr[i];
            u32 cp = (c + GSZ - 1) / GSZ * GSZ;
            s += c; sp += cp;
            hdr[9 + i]  = s;
            hdr[18 + i] = sp;
        }
        hdr[30] = sp;   // M_p
    }
}

// win2flat[j] = j + (bsi_p[b]-bsi[b]);  stored at d_out[M_p + j]
__global__ void k_win2flat(const int* __restrict__ coords, int n,
                           const u32* __restrict__ hdr, float* __restrict__ out) {
    int j = blockIdx.x * 256 + threadIdx.x;
    if (j >= n) return;
    u32 Mp = hdr[30];
    int b = coords[4 * j] & (NB - 1);
    u32 bias = hdr[17 + b] - hdr[8 + b];
    out[(size_t)Mp + j] = (float)((u32)j + bias);
}

// flat2win at d_out[0..Mp), mask at d_out[Mp+N .. Mp+N+Mp)
__global__ void k_flat2win(int n, const u32* __restrict__ hdr, float* __restrict__ out) {
    u32 Mp = hdr[30];
    u32 p = blockIdx.x * 256 + threadIdx.x;
    if (p >= Mp) return;
    int b = 0;
#pragma unroll
    for (int i = 0; i < NB - 1; ++i) if (p >= hdr[18 + i]) b = i + 1;
    u32 seg0 = hdr[17 + b], seg1 = hdr[18 + b];
    u32 c    = hdr[b];
    u32 cp   = seg1 - seg0;
    u32 bsi  = hdr[8 + b];
    u32 bias = seg0 - bsi;
    u32 local = p - seg0;
    u32 rem   = c % GSZ;
    u32 f     = p - bias;
    if (rem != 0) {
        u32 startLocal = cp - GSZ + rem;
        if (local >= startLocal) {
            if (cp != GSZ) f = p - GSZ - bias;                    // copy-from-previous-group
            else           f = bsi + ((local - startLocal) % c);  // tile branch
        }
    }
    out[p] = (float)f;
    out[(size_t)Mp + (u32)n + p] = (local >= c) ? 1.0f : 0.0f;
}

__global__ void k_keys(const int* __restrict__ coords, int n, const u32* __restrict__ hdr,
                       int axis, u32* __restrict__ keys, u32* __restrict__ idx) {
    int j = blockIdx.x * 256 + threadIdx.x;
    if (j >= n) return;
    u32 b = (u32)(coords[4 * j] & (NB - 1));
    u32 z = (u32)coords[4 * j + 1];
    u32 y = (u32)coords[4 * j + 2];
    u32 x = (u32)coords[4 * j + 3];
    u32 wcx = x / WX, wcy = y / WY, wcz = z / WZ;
    u32 cix = x % WX, ciy = y % WY, ciz = z % WZ;
    u32 mymz = hdr[26], mxmz = hdr[27], mz = hdr[28], mper = hdr[29];
    u32 key;
    if (axis == 0)
        key = (b * mper + wcx * mymz + wcy * mz + wcz) * (u32)VOL + cix * (WY * WZ) + ciy * WZ + ciz;
    else
        key = (b * mper + wcy * mxmz + wcx * mz + wcz) * (u32)VOL + ciy * (WX * WZ) + cix * WZ + ciz;
    keys[j] = key;
    idx[j]  = (u32)j;
}

// --- radix pass kernel 1: per-tile 256-bin histogram, digit-major global layout ---
__global__ void k_hist(const u32* __restrict__ keys, int n, int shift,
                       u32* __restrict__ hist, int numTiles) {
    __shared__ u32 h[256];
    h[threadIdx.x] = 0;
    __syncthreads();
    int w = threadIdx.x >> 5, lane = threadIdx.x & 31;
    int tileBase = blockIdx.x * TILE;
    // prefetch next tile (global_prefetch_b8)
    int pf = tileBase + TILE + (threadIdx.x << 3);
    if (pf < n) __builtin_prefetch(&keys[pf], 0, 0);
#pragma unroll
    for (int k = 0; k < 8; ++k) {
        int e = tileBase + w * 256 + k * 32 + lane;
        if (e < n) atomicAdd(&h[(keys[e] >> shift) & 255u], 1u);
    }
    __syncthreads();
    hist[(size_t)threadIdx.x * (u32)numTiles + blockIdx.x] = h[threadIdx.x];
}

// --- radix pass kernel 2a: WMMA tensor-core chunk scan (2048 elems / block, 256 / wave) ---
// Chunk is staged into LDS with the async copy engine (global_load_async_to_lds_b128,
// ASYNCcnt path), then each wave treats its 256 values as a 16x16 row-major matrix X
// and computes the row-inclusive scan Y = X * U (U upper-triangular ones) on the matrix
// unit. Inputs are per-(digit,tile) counts <= 2048 => exact in f16; f32 accum < 2^24.
__global__ void k_scan1(u32* __restrict__ data, int S, u32* __restrict__ chunkSums) {
    __shared__ __align__(16) u32 stage[TILE];
    __shared__ float rowTot[8][16];
    __shared__ float waveTot[8];
    int tid = threadIdx.x, w = tid >> 5, lane = tid & 31;
    int g0 = blockIdx.x * TILE;

    bool full = (g0 + TILE) <= S;      // uniform per block
    if (full) {
#pragma unroll
        for (int it = 0; it < 2; ++it) {
            int elem = it * 1024 + tid * 4;                  // 4 dwords = 16B per lane
            unsigned long long ga = (unsigned long long)(uintptr_t)(data + g0 + elem);
            u32 la = (u32)(uintptr_t)(lds3_u32*)&stage[elem];
            asm volatile("global_load_async_to_lds_b128 %0, %1, off"
                         :: "v"(la), "v"(ga) : "memory");
        }
        asm volatile("s_wait_asynccnt 0x0" ::: "memory");
    } else {
        for (int k = 0; k < 8; ++k) {
            int i = k * 256 + tid;
            int g = g0 + i;
            stage[i] = (g < S) ? data[g] : 0u;
        }
    }
    __syncthreads();

    const u32* bw = &stage[w * 256];
    int nm = lane & 15;                 // A: row m = lane%16 ; B/D: col n = lane%16
    int k0 = (lane < 16) ? 0 : 8;       // A halves 0..7 -> K = k0..k0+7 ; halves 8..15 -> K>=16 (zero)
    v16h a, bmat;
#pragma unroll
    for (int j = 0; j < 8; ++j) a[j] = (_Float16)(float)bw[nm * 16 + (k0 + j)];
#pragma unroll
    for (int j = 8; j < 16; ++j) a[j] = (_Float16)0.0f;
    // B (32x16): halves 0..7 -> K = k0..k0+7 ; halves 8..15 -> K>=16 (zero-fill)
#pragma unroll
    for (int j = 0; j < 16; ++j)
        bmat[j] = (j < 8) ? (_Float16)((k0 + j <= nm) ? 1.0f : 0.0f) : (_Float16)0.0f;

    v8f c = {0.f, 0.f, 0.f, 0.f, 0.f, 0.f, 0.f, 0.f};
    v8f d = __builtin_amdgcn_wmma_f32_16x16x32_f16(false, a, false, bmat,
                                                   (short)0, c, false, false);

    // D layout: lane n=lane%16, rows m = r + 8*(lane/16). Column 15 = row totals.
    int mBase = (lane >> 4) * 8;
    if (nm == 15) {
#pragma unroll
        for (int r = 0; r < 8; ++r) rowTot[w][mBase + r] = d[r];
    }
    __syncthreads();
    if (lane == 0) {
        float s = 0.f;
        for (int r = 0; r < 16; ++r) s += rowTot[w][r];
        waveTot[w] = s;
    }
    __syncthreads();

    float waveOff = 0.f;
    for (int w2 = 0; w2 < w; ++w2) waveOff += waveTot[w2];
    float rpre[16];
    {
        float s = 0.f;
        for (int i = 0; i < 16; ++i) { rpre[i] = s; s += rowTot[w][i]; }
    }
#pragma unroll
    for (int r = 0; r < 8; ++r) {
        int mm = mBase + r;
        int e  = mm * 16 + nm;
        int g  = g0 + w * 256 + e;
        if (g < S) {
            float excl = d[r] + rpre[mm] + waveOff - (float)stage[w * 256 + e];
            data[g] = (u32)(excl + 0.5f);
        }
    }
    if (tid == 0) {
        float s = 0.f;
        for (int i = 0; i < 8; ++i) s += waveTot[i];
        chunkSums[blockIdx.x] = (u32)(s + 0.5f);
    }
}

__global__ void k_scan2(u32* __restrict__ chunkSums, int numChunks) {
    if (threadIdx.x == 0 && blockIdx.x == 0) {
        u32 s = 0;
        for (int i = 0; i < numChunks; ++i) { u32 t = chunkSums[i]; chunkSums[i] = s; s += t; }
    }
}

__global__ void k_scan3(u32* __restrict__ data, int S, const u32* __restrict__ chunkSums) {
    u32 off = chunkSums[blockIdx.x];
    for (int k = 0; k < 8; ++k) {
        int g = blockIdx.x * TILE + k * 256 + threadIdx.x;
        if (g < S) data[g] += off;
    }
}

// --- radix pass kernel 3: stable scatter (wave-level multisplit via ballots) ---
__global__ void k_scatter(const u32* __restrict__ keysIn, const u32* __restrict__ idxIn,
                          u32* __restrict__ keysOut, u32* __restrict__ idxOut,
                          int n, int shift, const u32* __restrict__ hist, int numTiles) {
    __shared__ u32 digBase[256];
    __shared__ u32 wh[8][256];
    int tid = threadIdx.x, w = tid >> 5, lane = tid & 31;
    digBase[tid] = hist[(size_t)tid * (u32)numTiles + blockIdx.x];
    for (int i = tid; i < 8 * 256; i += 256) ((u32*)wh)[i] = 0;
    __syncthreads();

    int base = blockIdx.x * TILE + w * 256;   // wave-contiguous => stable order
    u32 keys[8], digs[8];
    bool val[8];
#pragma unroll
    for (int k = 0; k < 8; ++k) {
        int e = base + k * 32 + lane;
        val[k]  = e < n;
        keys[k] = val[k] ? keysIn[e] : 0u;
        digs[k] = (keys[k] >> shift) & 255u;
        if (val[k]) atomicAdd(&wh[w][digs[k]], 1u);
    }
    __syncthreads();
    // exclusive across waves, per digit
    {
        u32 s = 0;
        for (int w2 = 0; w2 < 8; ++w2) { u32 t = wh[w2][tid]; wh[w2][tid] = s; s += t; }
    }
    __syncthreads();

    u32 lmask = (1u << lane) - 1u;
    for (int k = 0; k < 8; ++k) {
        u32 dg = digs[k];
        unsigned mask = (unsigned)__ballot((int)val[k]);
#pragma unroll
        for (int bit = 0; bit < 8; ++bit) {
            unsigned bb = (unsigned)__ballot((int)((dg >> bit) & 1u));
            mask &= ((dg >> bit) & 1u) ? bb : ~bb;
        }
        if (val[k]) {
            u32 prior = (u32)__popc(mask & lmask);
            u32 cur   = wh[w][dg];
            u32 pos   = digBase[dg] + cur + prior;
            if (prior == 0) wh[w][dg] = cur + (u32)__popc(mask);   // leader advances counter
            keysOut[pos] = keys[k];
            idxOut[pos]  = idxIn[base + k * 32 + lane];
        }
    }
}

__global__ void k_write_idx(const u32* __restrict__ idx, int n,
                            const u32* __restrict__ hdr, float* __restrict__ out, int axis) {
    int j = blockIdx.x * 256 + threadIdx.x;
    if (j >= n) return;
    u32 Mp = hdr[30];
    size_t off = (size_t)2 * Mp + (size_t)n + (axis ? (size_t)n : 0);
    out[off + j] = (float)idx[j];
}

extern "C" void kernel_launch(void* const* d_in, const int* in_sizes, int n_in,
                              void* d_out, int out_size, void* d_ws, size_t ws_size,
                              hipStream_t stream) {
    const int* coords = (const int*)d_in[0];
    const int* sparse = (const int*)d_in[2];
    float* out = (float*)d_out;

    const int N       = in_sizes[0] / 4;
    const int MpMax   = N + NB * (GSZ - 1);
    const int numTiles  = (N + TILE - 1) / TILE;
    const int S         = 256 * numTiles;
    const int numChunks = (S + TILE - 1) / TILE;

    u32* ws32  = (u32*)d_ws;
    u32* hdr   = ws32;
    u32* kA    = ws32 + 64;
    u32* iA    = kA + N;
    u32* kB    = iA + N;
    u32* iB    = kB + N;
    u32* hist  = iB + N;   // byte offset 256 + 16N from 256B-aligned base => 16B aligned
    u32* chunk = hist + S;

    const int gN  = (N + 255) / 256;
    const int gMp = (MpMax + 255) / 256;

    k_setup<<<1, 1, 0, stream>>>(sparse, hdr);
    k_count<<<(gN < 2048 ? gN : 2048), 256, 0, stream>>>(coords, N, hdr);
    k_offsets<<<1, 1, 0, stream>>>(hdr);
    k_win2flat<<<gN, 256, 0, stream>>>(coords, N, hdr, out);
    k_flat2win<<<gMp, 256, 0, stream>>>(N, hdr, out);

    const int shifts[4] = {0, 8, 16, 24};
    for (int axis = 0; axis < 2; ++axis) {
        k_keys<<<gN, 256, 0, stream>>>(coords, N, hdr, axis, kA, iA);
        u32 *ki = kA, *ii = iA, *ko = kB, *io = iB;
        for (int p = 0; p < 4; ++p) {
            k_hist<<<numTiles, 256, 0, stream>>>(ki, N, shifts[p], hist, numTiles);
            k_scan1<<<numChunks, 256, 0, stream>>>(hist, S, chunk);
            k_scan2<<<1, 1, 0, stream>>>(chunk, numChunks);
            k_scan3<<<numChunks, 256, 0, stream>>>(hist, S, chunk);
            k_scatter<<<numTiles, 256, 0, stream>>>(ki, ii, ko, io, N, shifts[p], hist, numTiles);
            u32* t;
            t = ki; ki = ko; ko = t;
            t = ii; ii = io; io = t;
        }
        // 4 passes => results back in kA/iA
        k_write_idx<<<gN, 256, 0, stream>>>(ii, N, hdr, out, axis);
    }
}
